// TernaryConv1d_45243185496243
// MI455X (gfx1250) — matrix operand order repaired and checked by reading, not compiled, not run on hardware
//
#include <hip/hip_runtime.h>

// ---------- types ----------
typedef __bf16 bf16_t;
typedef __attribute__((ext_vector_type(16))) __bf16 v16bf;
typedef __attribute__((ext_vector_type(8)))  __bf16 v8bf;
typedef __attribute__((ext_vector_type(8)))  float  v8f;
typedef __attribute__((ext_vector_type(4)))  unsigned int v4u;
typedef __attribute__((ext_vector_type(8)))  int v8i;
typedef __attribute__((ext_vector_type(4)))  int v4i;

__device__ __forceinline__ bf16_t f2bf(float f) {
    union { float f; unsigned u; } v; v.f = f;
    unsigned r = v.u + 0x7FFFu + ((v.u >> 16) & 1u);   // round-to-nearest-even
    unsigned short h = (unsigned short)(r >> 16);
    bf16_t b;
    __builtin_memcpy(&b, &h, 2);
    return b;
}

// ---------- problem constants ----------
#define CIN   256
#define COUT  512
#define LLEN  4096
#define NBAT  32
#define KW    3
#define KDIM  (CIN * KW)      // 768, reordered as kk*256 + c

// ---------- tiling ----------
#define BM    128
#define BN    128
#define BK    32
#define LDB   40              // padded LDS row (bf16): 80B, 16B-aligned, conflict-free
#define LDA   40              // same padding for the A (weights) tile
#define BROWS 130             // BN + 2 halo rows (l0-1 .. l0+128)

// TDM availability (device pass only; host pass takes fallback which is plain C++)
#if defined(__gfx1250__) && __has_builtin(__builtin_amdgcn_tensor_load_to_lds)
#define HAS_TDM 1
#else
#define HAS_TDM 0
#endif

// =====================================================================
// Pre-pass: ternarize weights per output filter, write bf16 t reordered
// as tq[o][kk*256 + c], and f32 scales[o].
// =====================================================================
__global__ __launch_bounds__(256) void ternarize_kernel(
        const float* __restrict__ w, bf16_t* __restrict__ tq,
        float* __restrict__ scales) {
    __shared__ float red[256];
    const int o   = blockIdx.x;
    const int tid = threadIdx.x;
    const float* wrow = w + (size_t)o * KDIM;

    float m = 0.f;
    for (int idx = tid; idx < KDIM; idx += 256)
        m = fmaxf(m, fabsf(wrow[idx]));
    red[tid] = m; __syncthreads();
    for (int s = 128; s > 0; s >>= 1) {
        if (tid < s) red[tid] = fmaxf(red[tid], red[tid + s]);
        __syncthreads();
    }
    const float thresh = 0.7f * red[0];
    __syncthreads();

    float num = 0.f, den = 0.f;
    for (int idx = tid; idx < KDIM; idx += 256) {
        float wv = wrow[idx];
        float t  = (wv > thresh) ? 1.f : ((wv < -thresh) ? -1.f : 0.f);
        num += wv * t;
        den += t * t;
        int c  = idx / 3;
        int kk = idx - 3 * c;
        tq[(size_t)o * KDIM + kk * CIN + c] = f2bf(t);
    }
    red[tid] = num; __syncthreads();
    for (int s = 128; s > 0; s >>= 1) {
        if (tid < s) red[tid] += red[tid + s];
        __syncthreads();
    }
    num = red[0]; __syncthreads();
    red[tid] = den; __syncthreads();
    for (int s = 128; s > 0; s >>= 1) {
        if (tid < s) red[tid] += red[tid + s];
        __syncthreads();
    }
    den = red[0];
    if (tid == 0) scales[o] = (den == 0.f) ? 1.f : (num / den);
}

// =====================================================================
// TDM: stage a 3D weight tile [kk=3][o=128][c=32] bf16 into LDS with the
// hardware padding feature (16 dwords data + 4 dwords pad -> LDA=40 rows).
// global layout: elem(o,kk,c) = tq[o*768 + kk*256 + c]
//   dim0 = c (contig, 32), stride(o) = 768, stride(kk) = 256
// =====================================================================
#if HAS_TDM
__device__ __forceinline__ void tdm_load_A(const bf16_t* gsrc, unsigned lds_off) {
    unsigned long long ga = (unsigned long long)gsrc;
    v4u g0 = {0u, 0u, 0u, 0u};
    g0[0] = 1u;                                        // count=1 (valid user D#)
    g0[1] = lds_off;                                   // lds_addr
    g0[2] = (unsigned)(ga & 0xFFFFFFFFu);              // global_addr[31:0]
    g0[3] = (unsigned)((ga >> 32) & 0x01FFFFFFu) | 0x80000000u;  // ga[56:32], type=2

    v8i g1 = {0, 0, 0, 0, 0, 0, 0, 0};
    g1[0] = (1 << 16)        // data_size = 2 bytes
          | (1 << 20)        // pad_enable
          | (3 << 22)        // pad_interval: 16 dwords
          | (3 << 25);       // pad_amount:   4 dwords
    g1[1] = (BK & 0xFFFF) << 16;       // tensor_dim0[15:0]  = 32
    g1[2] = (BM & 0xFFFF) << 16;       // tensor_dim1[15:0]  = 128
    g1[3] = (BK & 0xFFFF) << 16;       // tile_dim0 = 32
    g1[4] = (BM & 0xFFFF) | (KW << 16);// tile_dim1 = 128, tile_dim2 = 3
    g1[5] = KDIM;                      // tensor_dim0_stride = 768 (o rows)
    g1[6] = (CIN & 0xFFFF) << 16;      // tensor_dim1_stride = 256 (kk planes)
    g1[7] = 0;

    v4i g2 = {KW, 0, 0, 0};            // tensor_dim2 = 3; rest unused
    v4i g3 = {0, 0, 0, 0};

#if __has_include(<hip/amd_detail/amd_gfx1250_TDM.h>)
    v8i gx = {0, 0, 0, 0, 0, 0, 0, 0};
    __builtin_amdgcn_tensor_load_to_lds(g0, g1, g2, g3, gx, 0);
#else
    __builtin_amdgcn_tensor_load_to_lds(g0, g1, g2, g3, 0);
#endif
}
#endif

// =====================================================================
// Implicit-GEMM conv: out[b,o,l] = scale[o] * sum_{kk,c} t[o,kk,c]*x[b,c,l+kk-1]
// One x tile (with halo rows) serves all 3 taps: tap shift = LDS row shift.
// =====================================================================
__global__ __launch_bounds__(256) void conv_wmma_kernel(
        const float* __restrict__ x, const bf16_t* __restrict__ tq,
        const float* __restrict__ scales, float* __restrict__ out) {
    __shared__ __attribute__((aligned(16))) bf16_t Bs[BROWS * LDB];     // x tile [l][c]
    __shared__ __attribute__((aligned(16))) bf16_t As[KW * BM * LDA];   // t tile [kk][o][c]

    const int tid  = threadIdx.x;
    const int lane = tid & 31;
    const int wave = tid >> 5;
    const int wm   = wave >> 1;          // 0..3 : M sub-block
    const int wn   = wave & 1;           // 0..1 : N sub-block
    const int mb   = blockIdx.y;         // 0..3
    const int nb   = blockIdx.x;         // 0..1023
    const int b    = nb >> 5;            // batch
    const int l0   = (nb & 31) * BN;     // tile origin along L

    const int lhalf = (lane < 16) ? 0 : 1;
    const int lmod  = lane & 15;
    const int m_base = mb * BM + wm * 32;

    // per-lane output scales (C/D layout: lane-half selects M+8)
    float sc[2][8];
#pragma unroll
    for (int i = 0; i < 2; ++i)
#pragma unroll
        for (int v = 0; v < 8; ++v)
            sc[i][v] = scales[m_base + i * 16 + v + lhalf * 8];

    const v8f vzero = {0.f, 0.f, 0.f, 0.f, 0.f, 0.f, 0.f, 0.f};
    v8f acc[2][4];
#pragma unroll
    for (int i = 0; i < 2; ++i)
#pragma unroll
        for (int j = 0; j < 4; ++j) acc[i][j] = vzero;

    // cooperative-loader mapping for the x tile: 32 c-rows x 8 l-chunks of 16
    const int r   = tid >> 3;            // 0..31
    const int sub = tid & 7;             // 0..7
    const int la  = l0 + sub * 16;       // aligned 16-float chunk origin

    for (int c0 = 0; c0 < CIN; c0 += BK) {
        // ---- fill x tile: rows 1..128 = l0..l0+127, row 0 / 129 = halo ----
        const float* xrow = x + ((size_t)b * CIN + (c0 + r)) * LLEN;
        float4 f0 = *(const float4*)(xrow + la);
        float4 f1 = *(const float4*)(xrow + la + 4);
        float4 f2 = *(const float4*)(xrow + la + 8);
        float4 f3 = *(const float4*)(xrow + la + 12);
        float vals[16];
        *(float4*)(&vals[0])  = f0;
        *(float4*)(&vals[4])  = f1;
        *(float4*)(&vals[8])  = f2;
        *(float4*)(&vals[12]) = f3;
#pragma unroll
        for (int t = 0; t < 16; ++t)
            Bs[(sub * 16 + t + 1) * LDB + r] = f2bf(vals[t]);
        if (sub == 0)
            Bs[0 * LDB + r] = f2bf((l0 > 0) ? xrow[l0 - 1] : 0.f);
        if (sub == 7)
            Bs[(BROWS - 1) * LDB + r] = f2bf((l0 + BN < LLEN) ? xrow[l0 + BN] : 0.f);

        // ---- stage A tile [3][128][32] into LDS ----
#if HAS_TDM
        if (wave == 0) {
            tdm_load_A(tq + (size_t)(mb * BM) * KDIM + c0,
                       (unsigned)(unsigned long long)(&As[0]));
            __builtin_amdgcn_s_wait_tensorcnt(0);
        }
#else
        {
            // fallback: cooperative 16B copies, same padded layout
#pragma unroll
            for (int it = 0; it < 6; ++it) {
                int q    = it * 256 + tid;          // 0..1535
                int kk   = q >> 9;
                int rem  = q & 511;
                int orow = rem >> 2;
                int part = rem & 3;
                const bf16_t* src = tq + (size_t)(mb * BM + orow) * KDIM
                                       + kk * CIN + c0 + part * 8;
                bf16_t* dst = &As[(kk * BM + orow) * LDA + part * 8];
                *(v8bf*)dst = *(const v8bf*)src;
            }
        }
#endif
        // prefetch next c-chunk of x (global_prefetch_b8)
        if (c0 + BK < CIN)
            __builtin_prefetch(x + ((size_t)b * CIN + (c0 + BK + r)) * LLEN + la, 0, 1);
        __syncthreads();

        // ---- 3 taps x 8 WMMAs over the shared tiles ----
#pragma unroll
        for (int kk = 0; kk < KW; ++kk) {
            // A fragments: per-lane 16B+16B LDS loads matching 16-bit A layout
            v16bf afrag[2];
#pragma unroll
            for (int i = 0; i < 2; ++i) {
                int lm = wm * 32 + i * 16 + lmod;
                const bf16_t* ap = &As[(kk * BM + lm) * LDA + lhalf * 8];
                v8bf alo = *(const v8bf*)ap;         // K: klo..klo+7
                v8bf ahi = *(const v8bf*)(ap + 16);  // K: klo+16..klo+23
                v16bf a;
#pragma unroll
                for (int t = 0; t < 8; ++t) { a[t] = alo[t]; a[t + 8] = ahi[t]; }
                afrag[i] = a;
            }
#pragma unroll
            for (int j = 0; j < 4; ++j) {
                int nl = wn * 64 + j * 16 + lmod;
                const bf16_t* bp = &Bs[(nl + kk) * LDB + lhalf * 8];
                v8bf blo = *(const v8bf*)bp;
                v8bf bhi = *(const v8bf*)(bp + 16);
                v16bf bfrag;
#pragma unroll
                for (int t = 0; t < 8; ++t) { bfrag[t] = blo[t]; bfrag[t + 8] = bhi[t]; }
#pragma unroll
                for (int i = 0; i < 2; ++i) {
                    acc[i][j] = __builtin_amdgcn_wmma_f32_16x16x32_bf16(
                        false, afrag[i], false, bfrag,
                        (short)0, acc[i][j], false, false);
                }
            }
        }
        __syncthreads();
    }

    // ---- epilogue: scale and store ----
#pragma unroll
    for (int i = 0; i < 2; ++i) {
#pragma unroll
        for (int j = 0; j < 4; ++j) {
            int nidx = l0 + wn * 64 + j * 16 + lmod;
#pragma unroll
            for (int v = 0; v < 8; ++v) {
                int orow = m_base + i * 16 + v + lhalf * 8;
                out[((size_t)b * COUT + orow) * LLEN + nidx] = acc[i][j][v] * sc[i][v];
            }
        }
    }
}

// =====================================================================
// launcher
// =====================================================================
extern "C" void kernel_launch(void* const* d_in, const int* in_sizes, int n_in,
                              void* d_out, int out_size, void* d_ws, size_t ws_size,
                              hipStream_t stream) {
    const float* x = (const float*)d_in[0];   // [32,256,4096]
    const float* w = (const float*)d_in[1];   // [512,256,3]
    float* out = (float*)d_out;               // [32,512,4096]

    bf16_t* tq    = (bf16_t*)d_ws;                                     // 512*768 bf16
    float*  scale = (float*)((char*)d_ws + (size_t)COUT * KDIM * 2);   // 512 f32

    ternarize_kernel<<<COUT, 256, 0, stream>>>(w, tq, scale);

    dim3 grid(NBAT * (LLEN / BN), COUT / BM);  // (1024, 4)
    conv_wmma_kernel<<<grid, 256, 0, stream>>>(x, tq, scale, out);
}
